// FourierConvLayer_31653908971680
// MI455X (gfx1250) — matrix operand order repaired and checked by reading, not compiled, not run on hardware
//
#include <hip/hip_runtime.h>
#include <hip/hip_bf16.h>

// ---------------------------------------------------------------------------
// FourierConvLayer for MI455X (gfx1250, wave32, WMMA)
//   kr(u,v)[c,o] = sum_{y,x<3} cos(2*pi*(u*y+v*x)/128) * F[y,x,c,o]   (ki==kr)
//   out_real = (real+imag) . kr ;  out_imag = (imag-real) . kr
//   then train-mode BatchNorm (eps=1e-3, biased var) + LeakyReLU(0.2)
//
// Per frequency position: [S(8x64);D(8x64)] x kr(64x128) -> 16x128, i.e. one
// WMMA M-tile; 8 waves each own a 16x16 Cout tile and build their B fragment
// (their kr columns) directly in registers with packed bf16 FMAs from a
// transposed, bank-padded filter image resident in LDS.
// ---------------------------------------------------------------------------

typedef __attribute__((ext_vector_type(2)))  __bf16 bf16x2;
typedef __attribute__((ext_vector_type(16))) __bf16 v16bf;
typedef __attribute__((ext_vector_type(8)))  float  v8f;

#define HW 128
#define CIN 64
#define COUT 128
#define NB 8
#define NPOS (HW * HW)                       // 16384 frequency positions
#define POS_PER_WG 32
#define NWG (NPOS / POS_PER_WG)              // 512 workgroups
#define HALF_OUT ((size_t)NB * NPOS * COUT)  // 16777216 elems per output half
#define CNT_PER_CH ((float)(NB * NPOS))      // 131072 samples per channel

#define FPAD 72                              // padded cin row (64+8) in bf16
#define FSLAB (128 * FPAD)                   // one tap slab: [cout][cin+pad]

__device__ __forceinline__ unsigned short f2bf(float f) {
  unsigned int u = __builtin_bit_cast(unsigned int, f);
  u += 0x7FFFu + ((u >> 16) & 1u);           // round-to-nearest-even
  return (unsigned short)(u >> 16);
}

__device__ __forceinline__ bf16x2 bfma2(bf16x2 a, bf16x2 b, bf16x2 c) {
  return a * b + c;                          // aim: v_pk_fma_bf16
}

__global__ void zero_stats_kernel(float* stats) {
  stats[threadIdx.x] = 0.0f;                 // 512 floats: sums + sumsqs
}

// ---------------------------------------------------------------------------
// Kernel 1: spectral GEMM via v_wmma_f32_16x16x32_bf16 + BN stats atomics.
// ---------------------------------------------------------------------------
__global__ void __launch_bounds__(256)
fourier_gemm_kernel(const float* __restrict__ xr, const float* __restrict__ xi,
                    const float* __restrict__ filt, float* __restrict__ out,
                    float* __restrict__ stats) {
  extern __shared__ unsigned char smem_raw[];
  unsigned short* sFT = (unsigned short*)smem_raw;   // [9][128][72] bf16 (o-major)
  unsigned short* sA  = sFT + 9 * FSLAB;             // [2][16][64] bf16 (dbl buf)
  float*          sCos = (float*)(sA + 2 * 16 * CIN);// [128] cos(2*pi*t/128)

  const int tid  = threadIdx.x;
  const int lane = tid & 31;
  const int wave = tid >> 5;

  // ---- stage filters fp32 -> bf16, transposed to [tap][cout][cin+pad]
  for (int i = tid; i < 9 * CIN * COUT; i += 256) {
    const int k = i >> 13;            // tap
    const int r = i & 8191;
    const int c = r >> 7;             // cin
    const int o = r & 127;            // cout (fastest in global -> coalesced)
    sFT[k * FSLAB + o * FPAD + c] = f2bf(filt[i]);
  }
  if (tid < 128) sCos[tid] = __cosf((float)tid * 0.04908738521234052f);

  float csum = 0.0f, csq = 0.0f;
  const int m    = lane & 15;               // A row / C column within tile
  const int kb   = (lane < 16) ? 0 : 16;    // A-matrix K base (K-contig per lane)
  const int kb2  = (lane < 16) ? 0 : 16;    // B-matrix K base (K-contig per lane)
  const int ocol = wave * 16 + m;           // this wave's Cout column

  const bf16x2 zero2 = __builtin_bit_cast(bf16x2, 0u);

  for (int p = 0; p < POS_PER_WG; ++p) {
    const int pos = blockIdx.x * POS_PER_WG + p;
    const int u = pos >> 7, v = pos & (HW - 1);

    // ---- A tile into LDS (double buffered): rows 0-7 = S, rows 8-15 = D
    {
      const int n  = tid >> 5;
      const int c0 = (tid & 31) * 2;
      const size_t ib = (((size_t)n * HW + u) * HW + v) * CIN + c0;
      float2 r  = *(const float2*)(xr + ib);
      float2 im = *(const float2*)(xi + ib);
      __builtin_prefetch(xr + ib + CIN, 0, 1);   // next position's row
      __builtin_prefetch(xi + ib + CIN, 0, 1);
      unsigned int* sA32 = (unsigned int*)(sA + (p & 1) * (16 * CIN));
      sA32[(n * CIN + c0) >> 1] =
          (unsigned int)f2bf(r.x + im.x) | ((unsigned int)f2bf(r.y + im.y) << 16);
      sA32[((n + 8) * CIN + c0) >> 1] =
          (unsigned int)f2bf(im.x - r.x) | ((unsigned int)f2bf(im.y - r.y) << 16);
    }
    __syncthreads();   // A writes visible; also guards read->overwrite (2 apart)

    // ---- 9 cosine tap weights, broadcast into bf16 pairs
    bf16x2 wp[9];
#pragma unroll
    for (int y = 0; y < 3; ++y)
#pragma unroll
      for (int x = 0; x < 3; ++x) {
        const unsigned short wb = f2bf(sCos[(u * y + v * x) & (HW - 1)]);
        wp[y * 3 + x] = __builtin_bit_cast(bf16x2, (unsigned int)wb * 0x10001u);
      }

    // ---- build this lane's B fragments (kr columns) in registers:
    //      b0 = K 0..31, b1 = K 32..63 (16 K-contiguous bf16 each, per lane)
    union Frag { v16bf v; bf16x2 pr[8]; uint4 q[2]; } a0, a1, b0, b1;
#pragma unroll
    for (int t = 0; t < 8; ++t) { b0.pr[t] = zero2; b1.pr[t] = zero2; }
#pragma unroll
    for (int k = 0; k < 9; ++k) {
      const unsigned short* fc = sFT + k * FSLAB + ocol * FPAD;
      const uint4 q0 = *(const uint4*)(fc + kb2);        // c = kb2 .. kb2+7
      const uint4 q1 = *(const uint4*)(fc + kb2 + 8);    // c = kb2+8 .. kb2+15
      const uint4 q2 = *(const uint4*)(fc + 32 + kb2);   // c = 32+kb2 ..
      const uint4 q3 = *(const uint4*)(fc + 40 + kb2);
      const bf16x2 w2 = wp[k];
      b0.pr[0] = bfma2(w2, __builtin_bit_cast(bf16x2, q0.x), b0.pr[0]);
      b0.pr[1] = bfma2(w2, __builtin_bit_cast(bf16x2, q0.y), b0.pr[1]);
      b0.pr[2] = bfma2(w2, __builtin_bit_cast(bf16x2, q0.z), b0.pr[2]);
      b0.pr[3] = bfma2(w2, __builtin_bit_cast(bf16x2, q0.w), b0.pr[3]);
      b0.pr[4] = bfma2(w2, __builtin_bit_cast(bf16x2, q1.x), b0.pr[4]);
      b0.pr[5] = bfma2(w2, __builtin_bit_cast(bf16x2, q1.y), b0.pr[5]);
      b0.pr[6] = bfma2(w2, __builtin_bit_cast(bf16x2, q1.z), b0.pr[6]);
      b0.pr[7] = bfma2(w2, __builtin_bit_cast(bf16x2, q1.w), b0.pr[7]);
      b1.pr[0] = bfma2(w2, __builtin_bit_cast(bf16x2, q2.x), b1.pr[0]);
      b1.pr[1] = bfma2(w2, __builtin_bit_cast(bf16x2, q2.y), b1.pr[1]);
      b1.pr[2] = bfma2(w2, __builtin_bit_cast(bf16x2, q2.z), b1.pr[2]);
      b1.pr[3] = bfma2(w2, __builtin_bit_cast(bf16x2, q2.w), b1.pr[3]);
      b1.pr[4] = bfma2(w2, __builtin_bit_cast(bf16x2, q3.x), b1.pr[4]);
      b1.pr[5] = bfma2(w2, __builtin_bit_cast(bf16x2, q3.y), b1.pr[5]);
      b1.pr[6] = bfma2(w2, __builtin_bit_cast(bf16x2, q3.z), b1.pr[6]);
      b1.pr[7] = bfma2(w2, __builtin_bit_cast(bf16x2, q3.w), b1.pr[7]);
    }

    // ---- A fragments (K-contiguous per lane half), 16B LDS loads
    const unsigned short* sAr = sA + (p & 1) * (16 * CIN);
    a0.q[0] = *(const uint4*)&sAr[m * CIN + kb];
    a0.q[1] = *(const uint4*)&sAr[m * CIN + 8 + kb];
    a1.q[0] = *(const uint4*)&sAr[m * CIN + 32 + kb];
    a1.q[1] = *(const uint4*)&sAr[m * CIN + 40 + kb];

    v8f acc = {};
    acc = __builtin_amdgcn_wmma_f32_16x16x32_bf16(false, a0.v, false, b0.v,
                                                  (short)0, acc, false, false);
    acc = __builtin_amdgcn_wmma_f32_16x16x32_bf16(false, a1.v, false, b1.v,
                                                  (short)0, acc, false, false);

    // ---- scatter C tile (lanes 0-15 -> real half, 16-31 -> imag half)
    const size_t obase = ((lane < 16) ? (size_t)0 : HALF_OUT)
                         + (size_t)pos * COUT + (size_t)ocol;
#pragma unroll
    for (int r = 0; r < 8; ++r) {
      float val = acc[r];                               // batch n = r
      out[obase + (size_t)r * ((size_t)NPOS * COUT)] = val;
      csum += val;
      csq = fmaf(val, val, csq);
    }
  }

  // ---- per-(half,channel) BN statistics
  const int ch = ((lane < 16) ? 0 : COUT) + ocol;
  atomicAdd(&stats[ch], csum);
  atomicAdd(&stats[256 + ch], csq);
}

// ---------------------------------------------------------------------------
// Kernel 2: in-place train-mode BatchNorm + LeakyReLU(0.2), float4 streaming.
// ---------------------------------------------------------------------------
__global__ void __launch_bounds__(256)
bn_leaky_kernel(float* __restrict__ out, const float* __restrict__ stats,
                const float* __restrict__ gr, const float* __restrict__ br,
                const float* __restrict__ gi, const float* __restrict__ bi) {
  __shared__ float sScale[256], sShift[256];
  const int t = threadIdx.x;
  {
    const int half = t >> 7, ch = t & 127;
    float mean = stats[half * 128 + ch] * (1.0f / CNT_PER_CH);
    float ex2  = stats[256 + half * 128 + ch] * (1.0f / CNT_PER_CH);
    float var  = fmaxf(ex2 - mean * mean, 0.0f);       // biased variance
    float g = half ? gi[ch] : gr[ch];
    float b = half ? bi[ch] : br[ch];
    float sc = g * rsqrtf(var + 1e-3f);
    sScale[t] = sc;
    sShift[t] = fmaf(-sc, mean, b);
  }
  __syncthreads();

  float4* o4 = (float4*)out;
  const long long total4 = (long long)(2 * HALF_OUT) / 4;   // 8388608
  for (long long i = (long long)blockIdx.x * blockDim.x + t; i < total4;
       i += (long long)gridDim.x * blockDim.x) {
    float4 x = o4[i];
    const int base = ((int)(i >> 22)) * 128 + (((int)i & 31) << 2);
    float y0 = fmaf(sScale[base + 0], x.x, sShift[base + 0]);
    float y1 = fmaf(sScale[base + 1], x.y, sShift[base + 1]);
    float y2 = fmaf(sScale[base + 2], x.z, sShift[base + 2]);
    float y3 = fmaf(sScale[base + 3], x.w, sShift[base + 3]);
    x.x = (y0 >= 0.0f) ? y0 : 0.2f * y0;
    x.y = (y1 >= 0.0f) ? y1 : 0.2f * y1;
    x.z = (y2 >= 0.0f) ? y2 : 0.2f * y2;
    x.w = (y3 >= 0.0f) ? y3 : 0.2f * y3;
    o4[i] = x;
  }
}

extern "C" void kernel_launch(void* const* d_in, const int* in_sizes, int n_in,
                              void* d_out, int out_size, void* d_ws, size_t ws_size,
                              hipStream_t stream) {
  const float* xr   = (const float*)d_in[0];  // inputs_real  [8,128,128,64]
  const float* xi   = (const float*)d_in[1];  // inputs_imag  [8,128,128,64]
  const float* filt = (const float*)d_in[2];  // filters      [3,3,64,128]
  const float* gr   = (const float*)d_in[3];
  const float* br   = (const float*)d_in[4];
  const float* gi   = (const float*)d_in[5];
  const float* bi   = (const float*)d_in[6];
  float* out   = (float*)d_out;               // [real half | imag half]
  float* stats = (float*)d_ws;                // 512 floats

  zero_stats_kernel<<<1, 512, 0, stream>>>(stats);

  const size_t lds_bytes =
      (size_t)(9 * FSLAB + 2 * 16 * CIN) * sizeof(unsigned short)
      + 128 * sizeof(float);                  // ~166 KB of the 320 KB WGP LDS
  fourier_gemm_kernel<<<NWG, 256, lds_bytes, stream>>>(xr, xi, filt, out, stats);

  bn_leaky_kernel<<<2048, 256, 0, stream>>>(out, stats, gr, br, gi, bi);
}
